// MCL_54889682043535
// MI455X (gfx1250) — matrix operand order repaired and checked by reading, not compiled, not run on hardware
//
#include <hip/hip_runtime.h>
#include <hip/hip_bf16.h>
#include <math.h>

// ---------------------------------------------------------------------------
// Problem constants (from reference)
// ---------------------------------------------------------------------------
#define MMOD  3
#define NSEQ  4096
#define DIN   1024
#define DH    2048
#define DP    256

#define TAU_INV   20.0f      // 1/0.05
#define LAMBDA_C  0.5f
#define EPS_C     1e-8f

// 128x128 block tile, K-step 32, 256 threads = 8 wave32 waves (4 x 2),
// each wave owns a 32x64 tile = 2x4 WMMA 16x16x32 bf16 fragments.
#define BT    128
#define KT    32
#define LDT   40             // padded LDS row stride in bf16 (80 B)
#define TILE  (BT * LDT)     // one LDS tile in bf16 elems
#define TILEB ((unsigned)(TILE * 2))  // tile size in bytes

typedef __attribute__((ext_vector_type(16))) __bf16 v16bf;
typedef __attribute__((ext_vector_type(8)))  __bf16 v8bf;
typedef __attribute__((ext_vector_type(4)))  __bf16 v4bf;
typedef __attribute__((ext_vector_type(2)))  __bf16 v2bf;
typedef __attribute__((ext_vector_type(8)))  float  v8f;

// ---------------------------------------------------------------------------
// CDNA5 async copy: global -> LDS, 16 B per lane, tracked by ASYNCcnt.
// Per-lane LDS destination (VDST) lets us apply the A-tile K-block map (identity here)
// for free. Low 32 bits of a generic LDS pointer are the wave-relative LDS
// address the hardware adds to LDS_BASE.
// ---------------------------------------------------------------------------
__device__ __forceinline__ void async_b128(unsigned lds_off,
                                           const __bf16* g) {
  asm volatile("global_load_async_to_lds_b128 %0, %1, off"
               :: "v"(lds_off), "v"((unsigned long long)(const void*)g)
               : "memory");
}
__device__ __forceinline__ unsigned lds_addr(const void* p) {
  return (unsigned)(unsigned long long)p;
}
__device__ __forceinline__ void wait_async4() {
  asm volatile("s_wait_asynccnt 0x4" ::: "memory");
}
__device__ __forceinline__ void wait_async0() {
  asm volatile("s_wait_asynccnt 0x0" ::: "memory");
}

// Load a 32-byte contiguous LDS region as one v16bf fragment (2 x b128).
__device__ __forceinline__ v16bf frag16(const __bf16* p) {
  v8bf lo = *(const v8bf*)p;
  v8bf hi = *(const v8bf*)(p + 8);
  return __builtin_shufflevector(lo, hi, 0, 1, 2, 3, 4, 5, 6, 7,
                                 8, 9, 10, 11, 12, 13, 14, 15);
}

// 8-block mapping for the A tile: identity, so a lane's A-fragment
// (K [0-15] or [16-31]) is contiguous, matching the B fragment order.
__device__ __forceinline__ int permblk(int cb) {
  return cb;
}

// ---------------------------------------------------------------------------
// Wave-level 32x64 step. LDS layouts (both row stride LDT):
//   As2[row][k]:    A rows (K contiguous, identity block map)
//   Bs2[col][k]:    B^T rows (K contiguous)
// Every fragment is a 32-byte contiguous LDS read.
// ---------------------------------------------------------------------------
__device__ __forceinline__ void wave_mma(const __bf16* As2, const __bf16* Bs2,
                                         int wr, int wc, int lane,
                                         v8f acc[2][4]) {
  const int koff = (lane >> 4) << 4;   // 0 or 16
  const int l15  = lane & 15;

  v16bf af[2];
#pragma unroll
  for (int r = 0; r < 2; ++r)
    af[r] = frag16(As2 + (wr * 32 + r * 16 + l15) * LDT + koff);

  v16bf bfr[4];
#pragma unroll
  for (int c = 0; c < 4; ++c)
    bfr[c] = frag16(Bs2 + (wc * 64 + c * 16 + l15) * LDT + koff);

#pragma unroll
  for (int r = 0; r < 2; ++r)
#pragma unroll
    for (int c = 0; c < 4; ++c)
      acc[r][c] = __builtin_amdgcn_wmma_f32_16x16x32_bf16(
          false, af[r], false, bfr[c], (short)0, acc[r][c], false, false);
}

// ---------------------------------------------------------------------------
// Pure-bf16 NT GEMM + bias + ReLU:  C[m] = relu(A[m] @ Bt[m]^T + bias[m])
// A bf16 row-major [Mrows x K]; Bt bf16 row-major [Ncols x K].
// LDS ping-pong double buffer fed by GLOBAL_LOAD_ASYNC_TO_LDS_B128.
// 4 async issues per thread per K-slab; in-order completion lets
// s_wait_asynccnt 4 retire exactly the previous slab. Last slab is peeled so
// the steady-state loop is branch-free; global pointers advance by KT.
// Optionally also writes a bf16 copy of C (for the next GEMM's A operand).
// ---------------------------------------------------------------------------
__global__ __launch_bounds__(256) void gemm_bias_relu_nt(
    const __bf16* __restrict__ A, const __bf16* __restrict__ Bt,
    const float* __restrict__ bias, float* __restrict__ C,
    __bf16* __restrict__ Cbf,
    int K, int Ncols, long sA, long sB, long sBias, long sC) {
  __shared__ __bf16 As2[2 * TILE];
  __shared__ __bf16 Bs2[2 * TILE];

  const int m = blockIdx.z;
  A    += (long)m * sA;
  Bt   += (long)m * sB;
  bias += (long)m * sBias;
  C    += (long)m * sC;
  if (Cbf) Cbf += (long)m * sC;

  const int row0 = blockIdx.y * BT;
  const int col0 = blockIdx.x * BT;
  const int tid  = threadIdx.x;
  const int lane = tid & 31;
  const int wave = tid >> 5;
  const int wr   = wave >> 1;
  const int wc   = wave & 1;

  // Per-thread staging slots (fixed for the whole kernel):
  //   A: two 8-elem blocks; B: one row-half (2 x 16B).
  const int ar0 = tid >> 2,         ac0 = tid & 3;
  const int ar1 = (tid + 256) >> 2, ac1 = tid & 3;   // same cb, +64 rows
  const int brr = tid >> 1,         bhh = (tid & 1) * 16;

  const __bf16* gA0 = A + (long)(row0 + ar0) * K + ac0 * 8;
  const __bf16* gA1 = A + (long)(row0 + ar1) * K + ac1 * 8;
  const __bf16* gB  = Bt + (long)(col0 + brr) * K + bhh;

  const unsigned lA0 = lds_addr(As2 + (ar0 * LDT + permblk(ac0) * 8));
  const unsigned lA1 = lds_addr(As2 + (ar1 * LDT + permblk(ac1) * 8));
  const unsigned lB  = lds_addr(Bs2 + (brr * LDT + bhh));

  // Stateful issue: loads current head positions, then advances by KT.
  auto issue = [&](int buf) {
    const unsigned boff = buf ? TILEB : 0u;
    async_b128(lA0 + boff, gA0);
    async_b128(lA1 + boff, gA1);
    async_b128(lB + boff,      gB);
    async_b128(lB + boff + 16, gB + 8);
    gA0 += KT; gA1 += KT; gB += KT;
  };

  v8f acc[2][4] = {};

  const int nt = K / KT;
  int buf = 0;
  issue(0);
  for (int kt = 0; kt < nt - 1; ++kt) {     // branch-free steady state
    issue(buf ^ 1);                         // prefetch next slab
    wait_async4();                          // retire current slab
    __syncthreads();                        // publish LDS across waves
    wave_mma(As2 + buf * TILE, Bs2 + buf * TILE, wr, wc, lane, acc);
    __syncthreads();                        // done reading before overwrite
    buf ^= 1;
  }
  wait_async0();                            // peeled last slab
  __syncthreads();
  wave_mma(As2 + buf * TILE, Bs2 + buf * TILE, wr, wc, lane, acc);

  // C/D frag layout: M = v + 8*(lane>>4), N = lane&15
  const int half = lane >> 4;
#pragma unroll
  for (int r = 0; r < 2; ++r)
#pragma unroll
    for (int c = 0; c < 4; ++c)
#pragma unroll
      for (int v = 0; v < 8; ++v) {
        int row = row0 + wr * 32 + r * 16 + v + 8 * half;
        int col = col0 + wc * 64 + c * 16 + (lane & 15);
        float val = fmaxf(acc[r][c][v] + bias[col], 0.0f);
        C[(long)row * Ncols + col] = val;
        if (Cbf) Cbf[(long)row * Ncols + col] = (__bf16)val;
      }
}

// ---------------------------------------------------------------------------
// 64x64 LDS transpose + fp32->bf16:  Wt[n][k] = (bf16) W[k][n]
// ---------------------------------------------------------------------------
__global__ __launch_bounds__(256) void transpose_cvt(
    const float* __restrict__ W, __bf16* __restrict__ Wt,
    int K, int N, long sW, long sWt) {
  __shared__ __bf16 t[64][72];
  const int m  = blockIdx.z;
  W  += (long)m * sW;
  Wt += (long)m * sWt;
  const int k0 = blockIdx.y * 64;
  const int n0 = blockIdx.x * 64;
  const int tid = threadIdx.x;
#pragma unroll
  for (int i = 0; i < 16; ++i) {
    int idx = tid + i * 256;
    int r = idx >> 6, c = idx & 63;
    t[r][c] = (__bf16)W[(long)(k0 + r) * N + (n0 + c)];
  }
  __syncthreads();
#pragma unroll
  for (int i = 0; i < 16; ++i) {
    int idx = tid + i * 256;
    int r = idx >> 6, c = idx & 63;          // r: n-local, c: k-local
    Wt[(long)(n0 + r) * K + (k0 + c)] = t[c][r];
  }
}

// ---------------------------------------------------------------------------
// Elementwise fp32 -> bf16 (vectorized x4)
// ---------------------------------------------------------------------------
__global__ __launch_bounds__(256) void cvt_bf16(const float* __restrict__ in,
                                                __bf16* __restrict__ outb,
                                                long n4) {
  long i = (long)blockIdx.x * 256 + threadIdx.x;
  if (i < n4) {
    float4 f = ((const float4*)in)[i];
    v4bf v;
    v[0] = (__bf16)f.x; v[1] = (__bf16)f.y;
    v[2] = (__bf16)f.z; v[3] = (__bf16)f.w;
    ((v4bf*)outb)[i] = v;
  }
}

// ---------------------------------------------------------------------------
// Row L2-normalize (clamped at EPS), fp32 in -> bf16 out. One block per row.
// ---------------------------------------------------------------------------
__global__ __launch_bounds__(128) void normalize_rows(
    const float* __restrict__ z, __bf16* __restrict__ zn) {
  const long row = blockIdx.x;
  const float* zr = z + row * DP;
  const int tid = threadIdx.x;
  float a = zr[2 * tid];
  float b = zr[2 * tid + 1];
  float s = a * a + b * b;
#pragma unroll
  for (int off = 16; off >= 1; off >>= 1) s += __shfl_xor(s, off, 32);
  __shared__ float wsum[4];
  if ((tid & 31) == 0) wsum[tid >> 5] = s;
  __syncthreads();
  float tot = wsum[0] + wsum[1] + wsum[2] + wsum[3];
  float inv = 1.0f / fmaxf(sqrtf(tot), EPS_C);
  v2bf v;
  v[0] = (__bf16)(a * inv);
  v[1] = (__bf16)(b * inv);
  *(v2bf*)(zn + row * DP + 2 * tid) = v;
}

// ---------------------------------------------------------------------------
// Pairwise contrastive kernel: cos = zn_a @ zn_b^T (NT WMMA, async-staged),
// E = exp(cos/tau) fused in registers; row sums via xor-butterfly within
// 16-lane halves (fixed M per half), column partials via xor-16; atomicAdd
// into per-pair rowsum/colsum; diagonal to ed. Pairs: 0->(0,1) 1->(0,2)
// 2->(1,2).
// ---------------------------------------------------------------------------
__global__ __launch_bounds__(256) void cos_pair_kernel(
    const __bf16* __restrict__ zn, float* __restrict__ rowsum,
    float* __restrict__ colsum, float* __restrict__ ed) {
  __shared__ __bf16 As2[2 * TILE];
  __shared__ __bf16 Bs2[2 * TILE];

  const int p  = blockIdx.z;
  const int ma = (p == 2) ? 1 : 0;
  const int mb = (p == 0) ? 1 : 2;
  const __bf16* Za = zn + (long)ma * NSEQ * DP;
  const __bf16* Zb = zn + (long)mb * NSEQ * DP;

  const int row0 = blockIdx.y * BT;
  const int col0 = blockIdx.x * BT;
  const int tid  = threadIdx.x;
  const int lane = tid & 31;
  const int wave = tid >> 5;
  const int wr   = wave >> 1;
  const int wc   = wave & 1;

  const int ar0 = tid >> 2,         ac0 = tid & 3;
  const int ar1 = (tid + 256) >> 2, ac1 = tid & 3;
  const int brr = tid >> 1,         bhh = (tid & 1) * 16;

  const __bf16* gA0 = Za + (long)(row0 + ar0) * DP + ac0 * 8;
  const __bf16* gA1 = Za + (long)(row0 + ar1) * DP + ac1 * 8;
  const __bf16* gB  = Zb + (long)(col0 + brr) * DP + bhh;

  const unsigned lA0 = lds_addr(As2 + (ar0 * LDT + permblk(ac0) * 8));
  const unsigned lA1 = lds_addr(As2 + (ar1 * LDT + permblk(ac1) * 8));
  const unsigned lB  = lds_addr(Bs2 + (brr * LDT + bhh));

  auto issue = [&](int buf) {
    const unsigned boff = buf ? TILEB : 0u;
    async_b128(lA0 + boff, gA0);
    async_b128(lA1 + boff, gA1);
    async_b128(lB + boff,      gB);
    async_b128(lB + boff + 16, gB + 8);
    gA0 += KT; gA1 += KT; gB += KT;
  };

  v8f acc[2][4] = {};

  const int nt = DP / KT;
  int buf = 0;
  issue(0);
  for (int kt = 0; kt < nt - 1; ++kt) {
    issue(buf ^ 1);
    wait_async4();
    __syncthreads();
    wave_mma(As2 + buf * TILE, Bs2 + buf * TILE, wr, wc, lane, acc);
    __syncthreads();
    buf ^= 1;
  }
  wait_async0();
  __syncthreads();
  wave_mma(As2 + buf * TILE, Bs2 + buf * TILE, wr, wc, lane, acc);

  const int half = lane >> 4;
  float csum[4] = {0.f, 0.f, 0.f, 0.f};

#pragma unroll
  for (int r = 0; r < 2; ++r)
#pragma unroll
    for (int v = 0; v < 8; ++v) {
      const int grow = row0 + wr * 32 + r * 16 + v + 8 * half;
      float rsum = 0.f;
#pragma unroll
      for (int c = 0; c < 4; ++c) {
        const int gcol = col0 + wc * 64 + c * 16 + (lane & 15);
        float e = __expf(acc[r][c][v] * TAU_INV);
        rsum += e;
        csum[c] += e;
        if (grow == gcol) ed[(long)p * NSEQ + grow] = e;
      }
#pragma unroll
      for (int off = 1; off < 16; off <<= 1) rsum += __shfl_xor(rsum, off, 32);
      if ((lane & 15) == 0)
        atomicAdd(&rowsum[(long)p * NSEQ + grow], rsum);
    }

#pragma unroll
  for (int c = 0; c < 4; ++c) {
    float cs = csum[c] + __shfl_xor(csum[c], 16, 32);
    if (lane < 16) {
      const int gcol = col0 + wc * 64 + c * 16 + lane;
      atomicAdd(&colsum[(long)p * NSEQ + gcol], cs);
    }
  }
}

// ---------------------------------------------------------------------------
// Init + final loss kernels
// ---------------------------------------------------------------------------
__global__ void zero_kernel(float* rowsum, float* colsum, float* ed,
                            float* loss_out) {
  int i = blockIdx.x * blockDim.x + threadIdx.x;
  if (i < 3 * NSEQ) {
    rowsum[i] = 0.f;
    colsum[i] = 0.f;
    ed[i]     = 0.f;
  }
  if (i == 0) *loss_out = 0.f;
}

__global__ __launch_bounds__(256) void loss_kernel(
    const float* __restrict__ rowsum, const float* __restrict__ colsum,
    const float* __restrict__ ed, float* __restrict__ loss_out) {
  const int p = blockIdx.x;
  const int tid = threadIdx.x;
  float acc = 0.f;
  for (int i = tid; i < NSEQ; i += 256) {
    float e  = ed[(long)p * NSEQ + i];
    float la = __logf((rowsum[(long)p * NSEQ + i] - e) / e);
    float lb = __logf((colsum[(long)p * NSEQ + i] - e) / e);
    acc += LAMBDA_C * la + (1.f - LAMBDA_C) * lb;
  }
#pragma unroll
  for (int off = 16; off >= 1; off >>= 1) acc += __shfl_xor(acc, off, 32);
  __shared__ float w[8];
  if ((tid & 31) == 0) w[tid >> 5] = acc;
  __syncthreads();
  if (tid == 0) {
    float t = 0.f;
    for (int i = 0; i < 8; ++i) t += w[i];
    atomicAdd(loss_out, t / (float)NSEQ);
  }
}

// ---------------------------------------------------------------------------
// Host-side launch
// ---------------------------------------------------------------------------
extern "C" void kernel_launch(void* const* d_in, const int* in_sizes, int n_in,
                              void* d_out, int out_size, void* d_ws,
                              size_t ws_size, hipStream_t stream) {
  const float* x  = (const float*)d_in[0];  // [3,4096,1024]
  const float* W1 = (const float*)d_in[1];  // [3,1024,2048]
  const float* b1 = (const float*)d_in[2];  // [3,2048]
  const float* W2 = (const float*)d_in[3];  // [3,2048,256]
  const float* b2 = (const float*)d_in[4];  // [3,256]

  float* out  = (float*)d_out;
  float* h    = out;                               // [3,4096,2048] fp32
  float* loss = out + (long)MMOD * NSEQ * DH;      // scalar (last element)

  // Workspace layout (all 16B-aligned):
  char* ws = (char*)d_ws;
  const size_t zB   = (size_t)MMOD * NSEQ * DP * sizeof(float);
  const size_t znB  = (size_t)MMOD * NSEQ * DP * sizeof(__bf16);
  const size_t w1tB = (size_t)MMOD * DH * DIN * sizeof(__bf16);
  const size_t w2tB = (size_t)MMOD * DP * DH * sizeof(__bf16);
  const size_t xbB  = (size_t)MMOD * NSEQ * DIN * sizeof(__bf16);
  const size_t hbB  = (size_t)MMOD * NSEQ * DH * sizeof(__bf16);

  float*  z      = (float*)ws;
  __bf16* zn     = (__bf16*)(ws + zB);
  __bf16* W1t    = (__bf16*)(ws + zB + znB);                  // [3, DH, DIN]
  __bf16* W2t    = (__bf16*)(ws + zB + znB + w1tB);           // [3, DP, DH]
  __bf16* xb     = (__bf16*)(ws + zB + znB + w1tB + w2tB);    // [3, NSEQ, DIN]
  __bf16* hb     = (__bf16*)(ws + zB + znB + w1tB + w2tB + xbB); // [3,NSEQ,DH]
  float*  rowsum = (float*)(ws + zB + znB + w1tB + w2tB + xbB + hbB);
  float*  colsum = rowsum + 3 * NSEQ;
  float*  ed     = colsum + 3 * NSEQ;

  zero_kernel<<<(3 * NSEQ + 255) / 256, 256, 0, stream>>>(rowsum, colsum, ed,
                                                          loss);

  // One-time operand prep (bandwidth passes): W1t/W2t transpose+cvt, x cvt.
  dim3 t1(DH / 64, DIN / 64, MMOD);
  transpose_cvt<<<t1, 256, 0, stream>>>(W1, W1t, DIN, DH,
                                        (long)DIN * DH, (long)DH * DIN);
  dim3 t2(DP / 64, DH / 64, MMOD);
  transpose_cvt<<<t2, 256, 0, stream>>>(W2, W2t, DH, DP,
                                        (long)DH * DP, (long)DP * DH);
  const long xn4 = (long)MMOD * NSEQ * DIN / 4;
  cvt_bf16<<<(unsigned)((xn4 + 255) / 256), 256, 0, stream>>>(x, xb, xn4);

  // h = relu(x @ W1 + b1)   (also emits bf16 copy hb for GEMM2's A side)
  dim3 g1(DH / BT, NSEQ / BT, MMOD);
  gemm_bias_relu_nt<<<g1, 256, 0, stream>>>(
      xb, W1t, b1, h, hb, DIN, DH,
      (long)NSEQ * DIN, (long)DH * DIN, (long)DH, (long)NSEQ * DH);

  // z = relu(h @ W2 + b2)
  dim3 g2(DP / BT, NSEQ / BT, MMOD);
  gemm_bias_relu_nt<<<g2, 256, 0, stream>>>(
      hb, W2t, b2, z, (__bf16*)nullptr, DH, DP,
      (long)NSEQ * DH, (long)DP * DH, (long)DP, (long)NSEQ * DP);

  normalize_rows<<<MMOD * NSEQ, 128, 0, stream>>>(z, zn);

  dim3 g3(NSEQ / BT, NSEQ / BT, 3);
  cos_pair_kernel<<<g3, 256, 0, stream>>>(zn, rowsum, colsum, ed);

  loss_kernel<<<3, 256, 0, stream>>>(rowsum, colsum, ed, loss);
}
